// MoEGEMMFused_11450382811965
// MI455X (gfx1250) — compile-verified, matrix operand order checked by reading, not run on hardware
//
#include <hip/hip_runtime.h>
#include <hip/hip_bf16.h>

// ---------------------------------------------------------------------------
// MoE (router + top2 softmax + grouped GEMM1/SwiGLU + grouped GEMM2 combine)
// MI455X / gfx1250: wave32, BF16 WMMA (v_wmma_f32_16x16x32_bf16),
// double-buffered LDS (1 barrier/stage), v_perm-based bf16 packing.
// ---------------------------------------------------------------------------

#define H_  2048
#define I_  1408
#define E_  8
#define K_  2
#define T_  8192          // B*S tokens
#define TK_ 16384         // T_*K_ routed pairs
#define BM_ 128
#define LDT 40            // LDS row stride (ushorts): 32 K elems + 8 pad (80B, 16B aligned)
#define MAXMT (TK_ / BM_ + E_)   // 136 max M-tiles across experts

typedef __attribute__((ext_vector_type(8)))  float          floatx8;
typedef __attribute__((ext_vector_type(16))) __bf16         bf16x16;
typedef __attribute__((ext_vector_type(8)))  unsigned short ushort8;
typedef __attribute__((ext_vector_type(4)))  unsigned int   uintx4;

// round-to-nearest-even scalar (epilogue only)
static __device__ __forceinline__ unsigned short f2bf(float f) {
    unsigned int u = __float_as_uint(f);
    u = (u + 0x7FFFu + ((u >> 16) & 1u)) >> 16;
    return (unsigned short)u;
}

// pack 2 floats -> 2 bf16 (round-half-up bias + single v_perm_b32)
static __device__ __forceinline__ unsigned int pack2bf(float a, float b) {
    unsigned int ua = __float_as_uint(a) + 0x8000u;
    unsigned int ub = __float_as_uint(b) + 0x8000u;
    return __builtin_amdgcn_perm(ub, ua, 0x07060302u);  // {ub.hi16, ua.hi16}
}

static __device__ __forceinline__ bf16x16 frag_cat(ushort8 a, ushort8 b) {
    union { struct { ushort8 lo, hi; } p; bf16x16 f; } u;
    u.p.lo = a; u.p.hi = b;
    return u.f;
}

// ---------------------------------------------------------------------------
// 1) Router: one wave per token -> top-2 experts + softmax weights
// ---------------------------------------------------------------------------
__global__ __launch_bounds__(256) void moe_router(
    const float* __restrict__ x, const float* __restrict__ rw,
    int* __restrict__ exp_id, float* __restrict__ exp_wt)
{
    const int lane  = threadIdx.x & 31;
    const int wave  = threadIdx.x >> 5;
    const int token = blockIdx.x * 8 + wave;
    const float4* xp = (const float4*)(x + (size_t)token * H_);

    float acc[E_];
#pragma unroll
    for (int e = 0; e < E_; ++e) acc[e] = 0.f;

#pragma unroll 4
    for (int i = 0; i < H_ / 128; ++i) {            // 16 iters, 128 floats / wave / iter
        float4 xv = xp[lane + i * 32];
#pragma unroll
        for (int e = 0; e < E_; ++e) {
            float4 wv = ((const float4*)(rw + e * H_))[lane + i * 32];
            acc[e] += xv.x * wv.x + xv.y * wv.y + xv.z * wv.z + xv.w * wv.w;
        }
    }
#pragma unroll
    for (int off = 16; off; off >>= 1)
#pragma unroll
        for (int e = 0; e < E_; ++e) acc[e] += __shfl_xor(acc[e], off, 32);

    if (lane == 0) {
        int b1 = 0;
#pragma unroll
        for (int e = 1; e < E_; ++e) if (acc[e] > acc[b1]) b1 = e;
        int b2 = (b1 == 0) ? 1 : 0;
#pragma unroll
        for (int e = 0; e < E_; ++e) if (e != b2 && e != b1 && acc[e] > acc[b2]) b2 = e;
        float e2 = __expf(acc[b2] - acc[b1]);
        float s  = 1.f / (1.f + e2);
        exp_id[token * 2 + 0] = b1;  exp_wt[token * 2 + 0] = s;
        exp_id[token * 2 + 1] = b2;  exp_wt[token * 2 + 1] = e2 * s;
    }
}

// ---------------------------------------------------------------------------
// 2) Deterministic counting sort by expert + M-tile table (single block)
// ---------------------------------------------------------------------------
__global__ __launch_bounds__(256) void moe_scan(
    const int* __restrict__ exp_id, const float* __restrict__ exp_wt,
    int* __restrict__ row_token, float* __restrict__ row_w,
    int* __restrict__ tile_expert, int* __restrict__ tile_row0,
    int* __restrict__ tile_rows, int* __restrict__ n_mtiles)
{
    __shared__ int hist[256][E_];
    const int tid = threadIdx.x;
    const int p0  = tid * (TK_ / 256);              // 64 pairs each

    int c[E_];
#pragma unroll
    for (int e = 0; e < E_; ++e) c[e] = 0;
    for (int i = 0; i < TK_ / 256; ++i) c[exp_id[p0 + i]]++;
#pragma unroll
    for (int e = 0; e < E_; ++e) hist[tid][e] = c[e];
    __syncthreads();

    if (tid == 0) {
        int tot[E_], base[E_];
        int run = 0;
        for (int e = 0; e < E_; ++e) {
            int s = 0;
            for (int t = 0; t < 256; ++t) s += hist[t][e];
            tot[e] = s; base[e] = run; run += s;
        }
        for (int e = 0; e < E_; ++e) {              // exclusive scan per expert
            int r = base[e];
            for (int t = 0; t < 256; ++t) { int v = hist[t][e]; hist[t][e] = r; r += v; }
        }
        int nm = 0;
        for (int e = 0; e < E_; ++e)
            for (int r = 0; r < tot[e]; r += BM_) {
                tile_expert[nm] = e;
                tile_row0[nm]   = base[e] + r;
                tile_rows[nm]   = (tot[e] - r < BM_) ? (tot[e] - r) : BM_;
                ++nm;
            }
        *n_mtiles = nm;
    }
    __syncthreads();

#pragma unroll
    for (int e = 0; e < E_; ++e) c[e] = hist[tid][e];
    for (int i = 0; i < TK_ / 256; ++i) {
        int p = p0 + i;
        int e = exp_id[p];
        int pos = c[e]++;
        row_token[pos] = p >> 1;                    // pair p came from token p/K
        row_w[pos]     = exp_wt[p];
    }
}

// ---------------------------------------------------------------------------
// 3) Gather sorted token activations -> BF16 buffer xs[TK_, H_]
// ---------------------------------------------------------------------------
__global__ __launch_bounds__(256) void moe_gather(
    const float* __restrict__ x, const int* __restrict__ row_token,
    unsigned short* __restrict__ xs)
{
    const int row   = blockIdx.x;
    const int token = row_token[row];
    const float4* src = (const float4*)(x + (size_t)token * H_) + threadIdx.x * 2;
    float4 a = src[0], b = src[1];
    uintx4 w;
    w[0] = pack2bf(a.x, a.y); w[1] = pack2bf(a.z, a.w);
    w[2] = pack2bf(b.x, b.y); w[3] = pack2bf(b.z, b.w);
    *(uintx4*)(xs + (size_t)row * H_ + threadIdx.x * 8) = w;
}

// ---------------------------------------------------------------------------
// 4) Zero output (exactly 2 atomic fp32 adds per element later -> deterministic)
// ---------------------------------------------------------------------------
__global__ __launch_bounds__(256) void moe_zero(float* __restrict__ out, int n4)
{
    float4 z = {0.f, 0.f, 0.f, 0.f};
    for (int i = blockIdx.x * blockDim.x + threadIdx.x; i < n4;
         i += gridDim.x * blockDim.x)
        ((float4*)out)[i] = z;
}

// ---------------------------------------------------------------------------
// 5) Grouped GEMM1 + fused SwiGLU:  xs[BM,K=H] x w1[e][2I,H] -> yact bf16
//    Block: 128 rows x (64 gate + 64 up paired features). 8 waves (4M x 2N).
// ---------------------------------------------------------------------------
__global__ __launch_bounds__(256) void moe_gemm1(
    const unsigned short* __restrict__ xs, const float* __restrict__ w1,
    unsigned short* __restrict__ yact,
    const int* __restrict__ tile_expert, const int* __restrict__ tile_row0,
    const int* __restrict__ tile_rows, const int* __restrict__ n_mtiles)
{
    const int t = blockIdx.x;
    if (t >= *n_mtiles) return;
    const int e     = tile_expert[t];
    const int row0  = tile_row0[t];
    const int mrows = tile_rows[t];
    const int n0h   = blockIdx.y * 64;              // feature offset in [0, I)
    const float* w1e = w1 + (size_t)e * (2 * I_) * H_;

    __shared__ __align__(16) unsigned short lA[2][BM_ * LDT];
    __shared__ __align__(16) unsigned short lB[2][BM_ * LDT];

    const int tid  = threadIdx.x;
    const int lane = tid & 31, wave = tid >> 5;
    const int wm = wave & 3, wn = wave >> 2;        // 4 x 2 wave grid
    const int hi = lane >> 4, lr = lane & 15;

    // per-thread staging coordinates (2 threads per row, 16 K-elems per seg)
    const int arow = tid >> 1, aseg = tid & 1;
    const int garow = row0 + (arow < mrows ? arow : 0);
    const unsigned short* aptr = xs + (size_t)garow * H_ + aseg * 16;
    const int wr = (arow < 64) ? (n0h + arow) : (I_ + n0h + (arow - 64));
    const float* bptr = w1e + (size_t)wr * H_ + aseg * 16;
    const int soff = arow * LDT + aseg * 16;

    floatx8 acc[2][4];
#pragma unroll
    for (int i = 0; i < 2; ++i)
#pragma unroll
        for (int j = 0; j < 4; ++j)
#pragma unroll
            for (int v = 0; v < 8; ++v) acc[i][j][v] = 0.f;

    // stage 0: global -> regs -> LDS buffer 0
    {
        ushort8 ra0 = *(const ushort8*)(aptr + 0);
        ushort8 ra1 = *(const ushort8*)(aptr + 8);
        float4 rb0 = ((const float4*)bptr)[0], rb1 = ((const float4*)bptr)[1];
        float4 rb2 = ((const float4*)bptr)[2], rb3 = ((const float4*)bptr)[3];
        *(ushort8*)&lA[0][soff + 0] = ra0;
        *(ushort8*)&lA[0][soff + 8] = ra1;
        uintx4 w0, w1v;
        w0[0]=pack2bf(rb0.x,rb0.y); w0[1]=pack2bf(rb0.z,rb0.w);
        w0[2]=pack2bf(rb1.x,rb1.y); w0[3]=pack2bf(rb1.z,rb1.w);
        w1v[0]=pack2bf(rb2.x,rb2.y); w1v[1]=pack2bf(rb2.z,rb2.w);
        w1v[2]=pack2bf(rb3.x,rb3.y); w1v[3]=pack2bf(rb3.z,rb3.w);
        *(uintx4*)&lB[0][soff + 0] = w0;
        *(uintx4*)&lB[0][soff + 8] = w1v;
    }
    __syncthreads();

    int cur = 0;
    for (int kc = 0; kc < H_; kc += 32) {
        const bool more = (kc + 32) < H_;
        ushort8 ra0, ra1; float4 rb0, rb1, rb2, rb3;
        if (more) {                                  // issue next K-slice loads
            aptr += 32; bptr += 32;
            ra0 = *(const ushort8*)(aptr + 0);
            ra1 = *(const ushort8*)(aptr + 8);
            rb0 = ((const float4*)bptr)[0]; rb1 = ((const float4*)bptr)[1];
            rb2 = ((const float4*)bptr)[2]; rb3 = ((const float4*)bptr)[3];
        }

        // compute from current buffer (loads above in flight)
        bf16x16 fa[2], fb[4];
#pragma unroll
        for (int i = 0; i < 2; ++i) {
            int r = wm * 32 + i * 16 + lr;
            fa[i] = frag_cat(*(const ushort8*)&lA[cur][r * LDT + hi * 8],
                             *(const ushort8*)&lA[cur][r * LDT + 16 + hi * 8]);
        }
#pragma unroll
        for (int j = 0; j < 4; ++j) {
            int n = ((j < 2) ? (wn * 32 + j * 16) : (64 + wn * 32 + (j - 2) * 16)) + lr;
            fb[j] = frag_cat(*(const ushort8*)&lB[cur][n * LDT + hi * 8],
                             *(const ushort8*)&lB[cur][n * LDT + 16 + hi * 8]);
        }
#pragma unroll
        for (int i = 0; i < 2; ++i)
#pragma unroll
            for (int j = 0; j < 4; ++j)
                acc[i][j] = __builtin_amdgcn_wmma_f32_16x16x32_bf16(
                    false, fa[i], false, fb[j], (short)0, acc[i][j], false, false);

        if (more) {                                  // stage into alternate buffer
            *(ushort8*)&lA[cur ^ 1][soff + 0] = ra0;
            *(ushort8*)&lA[cur ^ 1][soff + 8] = ra1;
            uintx4 w0, w1v;
            w0[0]=pack2bf(rb0.x,rb0.y); w0[1]=pack2bf(rb0.z,rb0.w);
            w0[2]=pack2bf(rb1.x,rb1.y); w0[3]=pack2bf(rb1.z,rb1.w);
            w1v[0]=pack2bf(rb2.x,rb2.y); w1v[1]=pack2bf(rb2.z,rb2.w);
            w1v[2]=pack2bf(rb3.x,rb3.y); w1v[3]=pack2bf(rb3.z,rb3.w);
            *(uintx4*)&lB[cur ^ 1][soff + 0] = w0;
            *(uintx4*)&lB[cur ^ 1][soff + 8] = w1v;
        }
        __syncthreads();
        cur ^= 1;
    }

    // Epilogue: SwiGLU in-register (gate tiles j=0,1 pair with up tiles j=2,3)
#pragma unroll
    for (int i = 0; i < 2; ++i) {
        const int rbase = wm * 32 + i * 16 + hi * 8;
#pragma unroll
        for (int j = 0; j < 2; ++j) {
            const int col = n0h + wn * 32 + j * 16 + lr;
#pragma unroll
            for (int v = 0; v < 8; ++v) {
                int r = rbase + v;
                if (r < mrows) {
                    float g = acc[i][j][v];
                    float a = g / (1.f + __expf(-g)) * acc[i][2 + j][v];
                    yact[(size_t)(row0 + r) * I_ + col] = f2bf(a);
                }
            }
        }
    }
}

// ---------------------------------------------------------------------------
// 6) Grouped GEMM2 + combine:  yact[BM,K=I] x w2[e][H,I] -> atomic f32 out
// ---------------------------------------------------------------------------
__global__ __launch_bounds__(256) void moe_gemm2(
    const unsigned short* __restrict__ yact, const float* __restrict__ w2,
    const int* __restrict__ row_token, const float* __restrict__ row_w,
    float* __restrict__ out,
    const int* __restrict__ tile_expert, const int* __restrict__ tile_row0,
    const int* __restrict__ tile_rows, const int* __restrict__ n_mtiles)
{
    const int t = blockIdx.x;
    if (t >= *n_mtiles) return;
    const int e     = tile_expert[t];
    const int row0  = tile_row0[t];
    const int mrows = tile_rows[t];
    const int n0    = blockIdx.y * 128;
    const float* w2e = w2 + (size_t)e * H_ * I_;

    __shared__ __align__(16) unsigned short lA[2][BM_ * LDT];
    __shared__ __align__(16) unsigned short lB[2][BM_ * LDT];

    const int tid  = threadIdx.x;
    const int lane = tid & 31, wave = tid >> 5;
    const int wm = wave & 3, wn = wave >> 2;
    const int hi = lane >> 4, lr = lane & 15;

    const int arow = tid >> 1, aseg = tid & 1;
    const int garow = row0 + (arow < mrows ? arow : 0);
    const unsigned short* aptr = yact + (size_t)garow * I_ + aseg * 16;
    const float* bptr = w2e + (size_t)(n0 + arow) * I_ + aseg * 16;
    const int soff = arow * LDT + aseg * 16;

    floatx8 acc[2][4];
#pragma unroll
    for (int i = 0; i < 2; ++i)
#pragma unroll
        for (int j = 0; j < 4; ++j)
#pragma unroll
            for (int v = 0; v < 8; ++v) acc[i][j][v] = 0.f;

    {
        ushort8 ra0 = *(const ushort8*)(aptr + 0);
        ushort8 ra1 = *(const ushort8*)(aptr + 8);
        float4 rb0 = ((const float4*)bptr)[0], rb1 = ((const float4*)bptr)[1];
        float4 rb2 = ((const float4*)bptr)[2], rb3 = ((const float4*)bptr)[3];
        *(ushort8*)&lA[0][soff + 0] = ra0;
        *(ushort8*)&lA[0][soff + 8] = ra1;
        uintx4 w0, w1v;
        w0[0]=pack2bf(rb0.x,rb0.y); w0[1]=pack2bf(rb0.z,rb0.w);
        w0[2]=pack2bf(rb1.x,rb1.y); w0[3]=pack2bf(rb1.z,rb1.w);
        w1v[0]=pack2bf(rb2.x,rb2.y); w1v[1]=pack2bf(rb2.z,rb2.w);
        w1v[2]=pack2bf(rb3.x,rb3.y); w1v[3]=pack2bf(rb3.z,rb3.w);
        *(uintx4*)&lB[0][soff + 0] = w0;
        *(uintx4*)&lB[0][soff + 8] = w1v;
    }
    __syncthreads();

    int cur = 0;
    for (int kc = 0; kc < I_; kc += 32) {
        const bool more = (kc + 32) < I_;
        ushort8 ra0, ra1; float4 rb0, rb1, rb2, rb3;
        if (more) {
            aptr += 32; bptr += 32;
            ra0 = *(const ushort8*)(aptr + 0);
            ra1 = *(const ushort8*)(aptr + 8);
            rb0 = ((const float4*)bptr)[0]; rb1 = ((const float4*)bptr)[1];
            rb2 = ((const float4*)bptr)[2]; rb3 = ((const float4*)bptr)[3];
        }

        bf16x16 fa[2], fb[4];
#pragma unroll
        for (int i = 0; i < 2; ++i) {
            int r = wm * 32 + i * 16 + lr;
            fa[i] = frag_cat(*(const ushort8*)&lA[cur][r * LDT + hi * 8],
                             *(const ushort8*)&lA[cur][r * LDT + 16 + hi * 8]);
        }
#pragma unroll
        for (int j = 0; j < 4; ++j) {
            int n = wn * 64 + j * 16 + lr;
            fb[j] = frag_cat(*(const ushort8*)&lB[cur][n * LDT + hi * 8],
                             *(const ushort8*)&lB[cur][n * LDT + 16 + hi * 8]);
        }
#pragma unroll
        for (int i = 0; i < 2; ++i)
#pragma unroll
            for (int j = 0; j < 4; ++j)
                acc[i][j] = __builtin_amdgcn_wmma_f32_16x16x32_bf16(
                    false, fa[i], false, fb[j], (short)0, acc[i][j], false, false);

        if (more) {
            *(ushort8*)&lA[cur ^ 1][soff + 0] = ra0;
            *(ushort8*)&lA[cur ^ 1][soff + 8] = ra1;
            uintx4 w0, w1v;
            w0[0]=pack2bf(rb0.x,rb0.y); w0[1]=pack2bf(rb0.z,rb0.w);
            w0[2]=pack2bf(rb1.x,rb1.y); w0[3]=pack2bf(rb1.z,rb1.w);
            w1v[0]=pack2bf(rb2.x,rb2.y); w1v[1]=pack2bf(rb2.z,rb2.w);
            w1v[2]=pack2bf(rb3.x,rb3.y); w1v[3]=pack2bf(rb3.z,rb3.w);
            *(uintx4*)&lB[cur ^ 1][soff + 0] = w0;
            *(uintx4*)&lB[cur ^ 1][soff + 8] = w1v;
        }
        __syncthreads();
        cur ^= 1;
    }

    // Epilogue: scale by routing weight, atomic-add combine (2 adds/element)
#pragma unroll
    for (int i = 0; i < 2; ++i) {
        const int rbase = wm * 32 + i * 16 + hi * 8;
#pragma unroll
        for (int v = 0; v < 8; ++v) {
            int r = rbase + v;
            if (r < mrows) {
                int grow   = row0 + r;
                int token  = row_token[grow];
                float wgt  = row_w[grow];
                float* orow = out + (size_t)token * H_;
#pragma unroll
                for (int j = 0; j < 4; ++j) {
                    int col = n0 + wn * 64 + j * 16 + lr;
                    __hip_atomic_fetch_add(&orow[col], acc[i][j][v] * wgt,
                                           __ATOMIC_RELAXED, __HIP_MEMORY_SCOPE_AGENT);
                }
            }
        }
    }
}

// ---------------------------------------------------------------------------
// Host launcher
// ---------------------------------------------------------------------------
extern "C" void kernel_launch(void* const* d_in, const int* in_sizes, int n_in,
                              void* d_out, int out_size, void* d_ws, size_t ws_size,
                              hipStream_t stream) {
    const float* x   = (const float*)d_in[0];   // [B,S,H]
    const float* rw  = (const float*)d_in[1];   // [E,H]
    const float* w1  = (const float*)d_in[2];   // [E,2I,H]
    const float* w2  = (const float*)d_in[3];   // [E,H,I]
    float* out = (float*)d_out;                 // [B,S,H] f32

    char* ws = (char*)d_ws;
    size_t o = 0;
    unsigned short* xs   = (unsigned short*)(ws + o); o += (size_t)TK_ * H_ * 2;  // 64 MB
    unsigned short* yact = (unsigned short*)(ws + o); o += (size_t)TK_ * I_ * 2;  // 44 MB
    int*   exp_id    = (int*)  (ws + o); o += TK_ * 4;
    float* exp_wt    = (float*)(ws + o); o += TK_ * 4;
    int*   row_token = (int*)  (ws + o); o += TK_ * 4;
    float* row_w     = (float*)(ws + o); o += TK_ * 4;
    int*   tile_expert = (int*)(ws + o); o += MAXMT * 4;
    int*   tile_row0   = (int*)(ws + o); o += MAXMT * 4;
    int*   tile_rows   = (int*)(ws + o); o += MAXMT * 4;
    int*   n_mtiles    = (int*)(ws + o); o += 4;

    moe_router<<<T_ / 8, 256, 0, stream>>>(x, rw, exp_id, exp_wt);
    moe_scan<<<1, 256, 0, stream>>>(exp_id, exp_wt, row_token, row_w,
                                    tile_expert, tile_row0, tile_rows, n_mtiles);
    moe_gather<<<TK_, 256, 0, stream>>>(x, row_token, xs);
    moe_zero<<<2048, 256, 0, stream>>>(out, (T_ * H_) / 4);
    moe_gemm1<<<dim3(MAXMT, I_ / 64), 256, 0, stream>>>(
        xs, w1, yact, tile_expert, tile_row0, tile_rows, n_mtiles);
    moe_gemm2<<<dim3(MAXMT, H_ / 128), 256, 0, stream>>>(
        yact, w2, row_token, row_w, out,
        tile_expert, tile_row0, tile_rows, n_mtiles);
}
